// catRBM_44616120271531
// MI455X (gfx1250) — compile-verified, matrix operand order checked by reading, not run on hardware
//
#include <hip/hip_runtime.h>
#include <hip/hip_bf16.h>

typedef __attribute__((ext_vector_type(16))) __bf16 v16bf;
typedef __attribute__((ext_vector_type(8)))  __bf16 v8bf;
typedef __attribute__((ext_vector_type(8)))  float  v8f;

#define B_SZ    256
#define H_SZ    4096
#define K_SZ    32768          // V * N = 1024 * 32
#define KSPLIT  8
#define KCHUNK  (K_SZ / KSPLIT) // 4096
#define BLOCK_N 128
#define KSTEP   32
#define NITER   (KCHUNK / KSTEP) // 128
#define PITCH   40               // bf16 elems per LDS row (32 + 8 pad, kills bank conflicts)
#define A_ELEMS (256 * PITCH)    // 10240
#define B_ELEMS (128 * PITCH)    // 5120
#define BUF_ELEMS (A_ELEMS + B_ELEMS) // 15360

__global__ void zero_f32(float* __restrict__ p, int n) {
    int i = blockIdx.x * blockDim.x + threadIdx.x;
    if (i < n) p[i] = 0.0f;
}

__device__ __forceinline__ v8bf cvt8(float4 x, float4 y) {
    v8bf r;
    r[0] = (__bf16)x.x; r[1] = (__bf16)x.y; r[2] = (__bf16)x.z; r[3] = (__bf16)x.w;
    r[4] = (__bf16)y.x; r[5] = (__bf16)y.y; r[6] = (__bf16)y.z; r[7] = (__bf16)y.w;
    return r;
}

// grid: (H/128, KSPLIT), block: 512 threads (16 wave32)
// Workgroup tile: 256(M) x 128(H) over a K-slab of 4096.
// Wave tile: 64(M) x 32(H)  -> 4x2 accumulator frags of 16x16 f32.
__global__ __launch_bounds__(512)
void gemm_bf16_wmma(const float* __restrict__ v, const float* __restrict__ W,
                    float* __restrict__ logits) {
    __shared__ __bf16 lds[BUF_ELEMS * 2];

    const int tid    = threadIdx.x;
    const int hbase  = blockIdx.x * BLOCK_N;
    const int k0base = blockIdx.y * KCHUNK;

    // ---- staging task mapping ----
    const int srow  = tid >> 1;      // A: 0..255 rows ; B: (tid<256) 0..127 rows
    const int shalf = tid & 1;       // k sub-half: floats [half*16, half*16+16)
    const float* aSrc = v + (size_t)srow * K_SZ + k0base + shalf * 16;
    const float* bSrc = W + (size_t)(hbase + srow) * K_SZ + k0base + shalf * 16;

    float4 a0, a1, a2, a3, b0, b1, b2, b3;
    a0=a1=a2=a3=b0=b1=b2=b3=make_float4(0.f,0.f,0.f,0.f);

    auto loadG = [&](int it) {
        const float4* ap = (const float4*)(aSrc + (size_t)it * KSTEP);
        a0 = ap[0]; a1 = ap[1]; a2 = ap[2]; a3 = ap[3];
        __builtin_prefetch(ap + 8, 0, 0);          // next k-step of this v row
        if (tid < 256) {
            const float4* bp = (const float4*)(bSrc + (size_t)it * KSTEP);
            b0 = bp[0]; b1 = bp[1]; b2 = bp[2]; b3 = bp[3];
            __builtin_prefetch(bp + 8, 0, 0);      // next k-step of this W row
        }
    };
    auto storeL = [&](int buf) {
        __bf16* base = lds + buf * BUF_ELEMS;
        *(v8bf*)(base + srow * PITCH + shalf * 16)     = cvt8(a0, a1);
        *(v8bf*)(base + srow * PITCH + shalf * 16 + 8) = cvt8(a2, a3);
        if (tid < 256) {
            __bf16* bb = base + A_ELEMS;
            *(v8bf*)(bb + srow * PITCH + shalf * 16)     = cvt8(b0, b1);
            *(v8bf*)(bb + srow * PITCH + shalf * 16 + 8) = cvt8(b2, b3);
        }
    };

    // ---- compute mapping (matches documented 16-bit A/B VGPR striping) ----
    const int lane  = tid & 31;
    const int wave  = tid >> 5;      // 0..15
    const int waveM = wave & 3;      // 4 groups of 64 rows
    const int waveN = wave >> 2;     // 4 groups of 32 cols
    const int lrow  = lane & 15;
    const int khalf = lane >> 4;     // lanes 16-31 take the other K sub-half

    v8f acc[4][2];
#pragma unroll
    for (int mi = 0; mi < 4; ++mi)
#pragma unroll
        for (int ni = 0; ni < 2; ++ni)
#pragma unroll
            for (int r = 0; r < 8; ++r) acc[mi][ni][r] = 0.0f;

    auto compute = [&](int buf) {
        const __bf16* base = lds + buf * BUF_ELEMS;
        v16bf Af[4], Bf[2];
#pragma unroll
        for (int mi = 0; mi < 4; ++mi) {
            const __bf16* p = base + (waveM * 64 + mi * 16 + lrow) * PITCH + khalf * 8;
            v8bf lo = *(const v8bf*)p;             // K = kb+0..7   (VGPRs 0-3)
            v8bf hi = *(const v8bf*)(p + 16);      // K = kb+16..23 (VGPRs 4-7)
#pragma unroll
            for (int j = 0; j < 8; ++j) { Af[mi][j] = lo[j]; Af[mi][j + 8] = hi[j]; }
        }
#pragma unroll
        for (int ni = 0; ni < 2; ++ni) {
            const __bf16* p = base + A_ELEMS + (waveN * 32 + ni * 16 + lrow) * PITCH + khalf * 8;
            v8bf lo = *(const v8bf*)p;
            v8bf hi = *(const v8bf*)(p + 16);
#pragma unroll
            for (int j = 0; j < 8; ++j) { Bf[ni][j] = lo[j]; Bf[ni][j + 8] = hi[j]; }
        }
#pragma unroll
        for (int mi = 0; mi < 4; ++mi)
#pragma unroll
            for (int ni = 0; ni < 2; ++ni)
                acc[mi][ni] = __builtin_amdgcn_wmma_f32_16x16x32_bf16(
                    false, Af[mi], false, Bf[ni], (short)0, acc[mi][ni], false, false);
    };

    // ---- double-buffered main loop ----
    loadG(0);
    storeL(0);
    __syncthreads();
#pragma unroll 1
    for (int it = 0; it < NITER; ++it) {
        if (it + 1 < NITER) loadG(it + 1);
        compute(it & 1);
        if (it + 1 < NITER) storeL((it + 1) & 1);
        __syncthreads();
    }

    // ---- epilogue: accumulate K-split partials ----
#pragma unroll
    for (int mi = 0; mi < 4; ++mi)
#pragma unroll
        for (int ni = 0; ni < 2; ++ni) {
            const int m0 = waveM * 64 + mi * 16 + khalf * 8;   // C layout: VGPR r -> M=r (+8 for hi lanes)
            const int h0 = hbase + waveN * 32 + ni * 16 + lrow; // lane&15 -> column
#pragma unroll
            for (int r = 0; r < 8; ++r)
                atomicAdd(&logits[(size_t)(m0 + r) * H_SZ + h0], acc[mi][ni][r]);
        }
}

// one block per batch row: softplus-sum over H, plus v[b] . b_v
__global__ __launch_bounds__(256)
void finalize_kernel(const float* __restrict__ v, const float* __restrict__ b_h,
                     const float* __restrict__ b_v, const float* __restrict__ logits,
                     float* __restrict__ out) {
    const int b = blockIdx.x;
    const int t = threadIdx.x;
    float acc = 0.0f;
    for (int h = t; h < H_SZ; h += 256) {
        float x = logits[(size_t)b * H_SZ + h] + b_h[h];
        acc += fmaxf(x, 0.0f) + log1pf(expf(-fabsf(x)));   // stable softplus
    }
    const float* vb = v + (size_t)b * K_SZ;
    float dot = 0.0f;
    for (int i = t; i < K_SZ; i += 256) dot = fmaf(vb[i], b_v[i], dot);
    acc += dot;

    __shared__ float red[256];
    red[t] = acc;
    __syncthreads();
    for (int s = 128; s > 0; s >>= 1) {
        if (t < s) red[t] += red[t + s];
        __syncthreads();
    }
    if (t == 0) out[b] = red[0];
}

extern "C" void kernel_launch(void* const* d_in, const int* in_sizes, int n_in,
                              void* d_out, int out_size, void* d_ws, size_t ws_size,
                              hipStream_t stream) {
    const float* v   = (const float*)d_in[0];
    const float* W   = (const float*)d_in[1];
    const float* b_h = (const float*)d_in[2];
    const float* b_v = (const float*)d_in[3];
    float* out    = (float*)d_out;
    float* logits = (float*)d_ws;              // B*H fp32 partial-sum buffer (4 MB)

    const int nlog = B_SZ * H_SZ;
    zero_f32<<<dim3((nlog + 255) / 256), dim3(256), 0, stream>>>(logits, nlog);
    gemm_bf16_wmma<<<dim3(H_SZ / BLOCK_N, KSPLIT), dim3(512), 0, stream>>>(v, W, logits);
    finalize_kernel<<<dim3(B_SZ), dim3(256), 0, stream>>>(v, b_h, b_v, logits, out);
}
